// VariationalGraohAutoEncoder_9045201126083
// MI455X (gfx1250) — matrix operand order repaired and checked by reading, not compiled, not run on hardware
//
#include <hip/hip_runtime.h>
#include <hip/hip_bf16.h>

typedef __attribute__((ext_vector_type(16))) _Float16 v16h;
typedef __attribute__((ext_vector_type(8)))  float    v8f;

// ================= prep kernels =================

__global__ void vgae_zero4(float4* __restrict__ p, long long n4) {
  long long stride = (long long)gridDim.x * blockDim.x;
  float4 z = {0.f, 0.f, 0.f, 0.f};
  for (long long i = (long long)blockIdx.x * blockDim.x + threadIdx.x; i < n4; i += stride)
    p[i] = z;
}

__global__ void vgae_idx32(const long long* __restrict__ s64, const long long* __restrict__ t64,
                           int* __restrict__ s32, int* __restrict__ t32, int E) {
  int stride = gridDim.x * blockDim.x;
  for (int i = blockIdx.x * blockDim.x + threadIdx.x; i < E; i += stride) {
    s32[i] = (int)s64[i];
    t32[i] = (int)t64[i];
  }
}

__global__ void vgae_degree(const int* __restrict__ dst, float* __restrict__ cnt, int E) {
  int stride = gridDim.x * blockDim.x;
  for (int i = blockIdx.x * blockDim.x + threadIdx.x; i < E; i += stride)
    atomicAdd(&cnt[dst[i]], 1.0f);
}

__global__ void vgae_scales(const float* __restrict__ cnt, float* __restrict__ sscale,
                            float* __restrict__ dinv, float* __restrict__ selfinv, int n) {
  int stride = gridDim.x * blockDim.x;
  for (int i = blockIdx.x * blockDim.x + threadIdx.x; i < n; i += stride) {
    float c = cnt[i];
    sscale[i]  = 1.0f / fmaxf(c, 1.0f);   // SAGE mean divisor
    float deg  = c + 1.0f;                // GCN degree incl. self loop
    dinv[i]    = rsqrtf(deg);
    selfinv[i] = 1.0f / deg;
  }
}

// per-edge GCN normalization weight
__global__ void vgae_wnorm(const int* __restrict__ src, const int* __restrict__ dst,
                           const float* __restrict__ dinv, float* __restrict__ wn, int E) {
  int stride = gridDim.x * blockDim.x;
  for (int i = blockIdx.x * blockDim.x + threadIdx.x; i < E; i += stride)
    wn[i] = dinv[src[i]] * dinv[dst[i]];
}

// W[din][dout] (f32, row-major) -> Wt[dout][din] (f16): per-lane-contiguous B fragments
__global__ void vgae_wprep(const float* __restrict__ W, _Float16* __restrict__ Wt,
                           int din, int dout) {
  int total = din * dout;
  int stride = gridDim.x * blockDim.x;
  for (int j = blockIdx.x * blockDim.x + threadIdx.x; j < total; j += stride) {
    int n = j / din;
    int k = j - n * din;
    Wt[j] = (_Float16)W[k * dout + n];
  }
}

// ================= edge gather/scatter (float4 gathers, f32 atomics into L2) =================

// agg[dst] += x[src]; feature dim = 4<<d4shift
__global__ void vgae_scatter_sum4(const float4* __restrict__ x4,
                                  const int* __restrict__ src, const int* __restrict__ dst,
                                  float* __restrict__ agg, int E, int d4shift) {
  long long total = (long long)E << d4shift;
  int dm = (1 << d4shift) - 1;
  long long stride = (long long)gridDim.x * blockDim.x;
  for (long long i = (long long)blockIdx.x * blockDim.x + threadIdx.x; i < total; i += stride) {
    int e = (int)(i >> d4shift);
    int q = (int)i & dm;
    int s = src[e];
    int t = dst[e];
    float4 v = x4[((long long)s << d4shift) + q];
    float* ap = agg + ((((long long)t << d4shift) + q) << 2);
    atomicAdd(ap + 0, v.x);
    atomicAdd(ap + 1, v.y);
    atomicAdd(ap + 2, v.z);
    atomicAdd(ap + 3, v.w);
  }
}

// GCN self-loop init: out = (1/deg)[row]*xw + bias  (d=64 -> 16 float4/row; also un-poisons d_out)
__global__ void vgae_gcn_init4(const float4* __restrict__ xw4, const float* __restrict__ selfinv,
                               const float4* __restrict__ bias4, float4* __restrict__ out4,
                               long long n4) {
  long long stride = (long long)gridDim.x * blockDim.x;
  for (long long i = (long long)blockIdx.x * blockDim.x + threadIdx.x; i < n4; i += stride) {
    float s = selfinv[i >> 4];
    float4 v = xw4[i];
    float4 b = bias4[(int)i & 15];
    float4 r = {s * v.x + b.x, s * v.y + b.y, s * v.z + b.z, s * v.w + b.w};
    out4[i] = r;
  }
}

// out[dst] += wnorm[e]*xw[src]  (d=64 -> 16 float4/edge)
__global__ void vgae_gcn_scatter4(const float4* __restrict__ xw4,
                                  const int* __restrict__ src, const int* __restrict__ dst,
                                  const float* __restrict__ wn,
                                  float* __restrict__ out, int E) {
  long long total = (long long)E << 4;
  long long stride = (long long)gridDim.x * blockDim.x;
  for (long long i = (long long)blockIdx.x * blockDim.x + threadIdx.x; i < total; i += stride) {
    int e = (int)(i >> 4);
    int q = (int)i & 15;
    int s = src[e];
    int t = dst[e];
    float w = wn[e];
    float4 v = xw4[((long long)s << 4) + q];
    float* ap = out + ((((long long)t << 4) + q) << 2);
    atomicAdd(ap + 0, w * v.x);
    atomicAdd(ap + 1, w * v.y);
    atomicAdd(ap + 2, w * v.z);
    atomicAdd(ap + 3, w * v.w);
  }
}

// ================= WMMA GEMM =================
// Fragment layouts used here (wave32; A and B share contiguous k order, not the ISA table order):
//   A 16x32 f16 : lane l -> row m=l&15; e -> k=16*(l>>4)+e
//                 (one contiguous 64B run per lane -> 4x b128 loads)
//   B 32x16 f16 : lane l -> col n=l&15; e -> k=16*(l>>4)+e (contiguous 32B in Wt[dout][din])
//   C 16x16 f32 : lane l -> col n=l&15; vgpr r -> row m=8*(l>>4)+r
__device__ __forceinline__ void vgae_gemm_accum(const float* __restrict__ A,
                                                const _Float16* __restrict__ Wt,
                                                float rs, int mrow, int n0,
                                                int din, int half, int mn, v8f cacc[4]) {
  const float* arow = A + (long long)mrow * din + (half << 4);
  for (int kk = 0; kk < din; kk += 32) {
    float4 f0 = *(const float4*)(arow + kk);
    float4 f1 = *(const float4*)(arow + kk + 4);
    float4 f2 = *(const float4*)(arow + kk + 8);
    float4 f3 = *(const float4*)(arow + kk + 12);
    v16h a;
    a[0]  = (_Float16)(f0.x * rs); a[1]  = (_Float16)(f0.y * rs);
    a[2]  = (_Float16)(f0.z * rs); a[3]  = (_Float16)(f0.w * rs);
    a[4]  = (_Float16)(f1.x * rs); a[5]  = (_Float16)(f1.y * rs);
    a[6]  = (_Float16)(f1.z * rs); a[7]  = (_Float16)(f1.w * rs);
    a[8]  = (_Float16)(f2.x * rs); a[9]  = (_Float16)(f2.y * rs);
    a[10] = (_Float16)(f2.z * rs); a[11] = (_Float16)(f2.w * rs);
    a[12] = (_Float16)(f3.x * rs); a[13] = (_Float16)(f3.y * rs);
    a[14] = (_Float16)(f3.z * rs); a[15] = (_Float16)(f3.w * rs);
#pragma unroll
    for (int t = 0; t < 4; ++t) {
      int ncol = n0 + (t << 4) + mn;
      v16h b = *(const v16h*)(Wt + (long long)ncol * din + kk + (half << 4));
      cacc[t] = __builtin_amdgcn_wmma_f32_16x16x32_f16(
          /*neg_a=*/false, a, /*neg_b=*/false, b,
          /*c_mod=*/(short)0, cacc[t], /*reuse_a=*/false, /*reuse_b=*/false);
    }
  }
}

// out[M,dout] = act( (rowscale .* A1) @ W1 [+ A2 @ W2] [+ bias] )
// W passed pre-transposed f16 [dout][din]. One wave = 16 rows x 64 cols.
// Requires M%16==0, din%32==0, dout%64==0.
__global__ void vgae_gemm_wmma(const float* __restrict__ A1, const _Float16* __restrict__ W1t,
                               const float* __restrict__ rowscale,
                               const float* __restrict__ A2, const _Float16* __restrict__ W2t,
                               const float* __restrict__ bias, float* __restrict__ out,
                               int M, int din, int dout, int relu) {
  int lane = threadIdx.x;                                  // blockDim.x == 32
  int rowTile = blockIdx.y * blockDim.y + threadIdx.y;     // wave-uniform
  int nRowTiles = M >> 4;
  if (rowTile >= nRowTiles) return;                        // whole wave exits -> EXEC all-1s
  int m0 = rowTile << 4;
  int n0 = blockIdx.x << 6;

  int half = lane >> 4;
  int mn   = lane & 15;
  int mrow = m0 + mn;

  v8f cacc[4] = {};

  float rs = rowscale ? rowscale[mrow] : 1.0f;
  vgae_gemm_accum(A1, W1t, rs, mrow, n0, din, half, mn, cacc);
  if (A2)                                                   // wave-uniform branch
    vgae_gemm_accum(A2, W2t, 1.0f, mrow, n0, din, half, mn, cacc);

#pragma unroll
  for (int t = 0; t < 4; ++t) {
    int ncol = n0 + (t << 4) + mn;
    float bv = bias ? bias[ncol] : 0.0f;
#pragma unroll
    for (int r = 0; r < 8; ++r) {
      int row = m0 + (half << 3) + r;
      float val = cacc[t][r] + bv;
      if (relu) val = fmaxf(val, 0.0f);
      out[(long long)row * dout + ncol] = val;
    }
  }
}

// ================= host launcher =================

static inline int vgae_nblk(long long total, int bs) {
  long long b = (total + bs - 1) / bs;
  if (b > 1048576) b = 1048576;
  if (b < 1) b = 1;
  return (int)b;
}

extern "C" void kernel_launch(void* const* d_in, const int* in_sizes, int n_in,
                              void* d_out, int out_size, void* d_ws, size_t ws_size,
                              hipStream_t stream) {
  const float*     x    = (const float*)d_in[0];
  const long long* ei   = (const long long*)d_in[1];
  const float*     W1l  = (const float*)d_in[2];
  const float*     W1r  = (const float*)d_in[3];
  const float*     b1   = (const float*)d_in[4];
  const float*     W2l  = (const float*)d_in[5];
  const float*     W2r  = (const float*)d_in[6];
  const float*     b2   = (const float*)d_in[7];
  const float*     Wmu  = (const float*)d_in[8];
  const float*     bmu  = (const float*)d_in[9];
  const float*     Wls  = (const float*)d_in[10];
  const float*     bls  = (const float*)d_in[11];

  const int d_in0 = 128, h0 = 256, h1 = 128, d_o = 64;
  const int N = in_sizes[0] / d_in0;
  const int E = in_sizes[1] / 2;
  const long long* src64 = ei;       // edge_index[0]
  const long long* dst64 = ei + E;   // edge_index[1]

  // ---- workspace layout (floats; all offsets multiples of 16 floats -> 32B aligned) ----
  float* ws      = (float*)d_ws;
  float* cnt     = ws;                              // N
  float* sscale  = cnt + N;                         // N
  float* dinv    = sscale + N;                      // N
  float* selfinv = dinv + N;                        // N
  float* agg     = selfinv + N;                     // N*256 (layer1 uses N*128)
  float* h1b     = agg + (long long)N * 256;        // N*256
  float* h2b     = h1b + (long long)N * 256;        // N*128
  float* xw      = agg;                             // N*64, reuse (agg dead after h2)
  float* wnorm   = h2b + (long long)N * 128;        // E
  int*   src32   = (int*)(wnorm + E);               // E
  int*   dst32   = src32 + E;                       // E
  _Float16* w1lt = (_Float16*)(dst32 + E);          // 128*256
  _Float16* w1rt = w1lt + d_in0 * h0;               // 128*256
  _Float16* w2lt = w1rt + d_in0 * h0;               // 256*128
  _Float16* w2rt = w2lt + h0 * h1;                  // 256*128
  _Float16* wmut = w2rt + h0 * h1;                  // 128*64
  _Float16* wlst = wmut + h1 * d_o;                 // 128*64

  float* out_mu = (float*)d_out;
  float* out_ls = out_mu + (long long)N * d_o;

  const int BS = 256;
  dim3 gblock(32, 4);  // 4 waves/block, 1 wave = 16x64 output tile
  int nRowTiles = N / 16;
  int gy = (nRowTiles + 3) / 4;

  // ---- index conversion, degrees, normalizers, weight prep ----
  vgae_idx32<<<vgae_nblk(E, BS), BS, 0, stream>>>(src64, dst64, src32, dst32, E);
  vgae_zero4<<<vgae_nblk(N / 4, BS), BS, 0, stream>>>((float4*)cnt, N / 4);
  vgae_degree<<<vgae_nblk(E, BS), BS, 0, stream>>>(dst32, cnt, E);
  vgae_scales<<<vgae_nblk(N, BS), BS, 0, stream>>>(cnt, sscale, dinv, selfinv, N);
  vgae_wnorm<<<vgae_nblk(E, BS), BS, 0, stream>>>(src32, dst32, dinv, wnorm, E);
  vgae_wprep<<<vgae_nblk(d_in0 * h0, BS), BS, 0, stream>>>(W1l, w1lt, d_in0, h0);
  vgae_wprep<<<vgae_nblk(d_in0 * h0, BS), BS, 0, stream>>>(W1r, w1rt, d_in0, h0);
  vgae_wprep<<<vgae_nblk(h0 * h1, BS), BS, 0, stream>>>(W2l, w2lt, h0, h1);
  vgae_wprep<<<vgae_nblk(h0 * h1, BS), BS, 0, stream>>>(W2r, w2rt, h0, h1);
  vgae_wprep<<<vgae_nblk(h1 * d_o, BS), BS, 0, stream>>>(Wmu, wmut, h1, d_o);
  vgae_wprep<<<vgae_nblk(h1 * d_o, BS), BS, 0, stream>>>(Wls, wlst, h1, d_o);

  // ---- SAGE layer 1: h1 = relu(mean(x[src]->dst) @ W1l + x @ W1r + b1) ----
  vgae_zero4<<<vgae_nblk((long long)N * d_in0 / 4, BS), BS, 0, stream>>>(
      (float4*)agg, (long long)N * d_in0 / 4);
  vgae_scatter_sum4<<<vgae_nblk((long long)E * (d_in0 / 4), BS), BS, 0, stream>>>(
      (const float4*)x, src32, dst32, agg, E, 5);
  {
    dim3 grid(h0 / 64, gy);
    vgae_gemm_wmma<<<grid, gblock, 0, stream>>>(agg, w1lt, sscale, x, w1rt, b1, h1b,
                                                N, d_in0, h0, 1);
  }

  // ---- SAGE layer 2: h2 = relu(mean(h1[src]->dst) @ W2l + h1 @ W2r + b2) ----
  vgae_zero4<<<vgae_nblk((long long)N * h0 / 4, BS), BS, 0, stream>>>(
      (float4*)agg, (long long)N * h0 / 4);
  vgae_scatter_sum4<<<vgae_nblk((long long)E * (h0 / 4), BS), BS, 0, stream>>>(
      (const float4*)h1b, src32, dst32, agg, E, 6);
  {
    dim3 grid(h1 / 64, gy);
    vgae_gemm_wmma<<<grid, gblock, 0, stream>>>(agg, w2lt, sscale, h1b, w2rt, b2, h2b,
                                                N, h0, h1, 1);
  }

  // ---- GCN head (mu) ----
  {
    dim3 grid(d_o / 64, gy);
    vgae_gemm_wmma<<<grid, gblock, 0, stream>>>(h2b, wmut, nullptr, nullptr, nullptr,
                                                nullptr, xw, N, h1, d_o, 0);
  }
  vgae_gcn_init4<<<vgae_nblk((long long)N * 16, BS), BS, 0, stream>>>(
      (const float4*)xw, selfinv, (const float4*)bmu, (float4*)out_mu, (long long)N * 16);
  vgae_gcn_scatter4<<<vgae_nblk((long long)E * 16, BS), BS, 0, stream>>>(
      (const float4*)xw, src32, dst32, wnorm, out_mu, E);

  // ---- GCN head (logstd) ----
  {
    dim3 grid(d_o / 64, gy);
    vgae_gemm_wmma<<<grid, gblock, 0, stream>>>(h2b, wlst, nullptr, nullptr, nullptr,
                                                nullptr, xw, N, h1, d_o, 0);
  }
  vgae_gcn_init4<<<vgae_nblk((long long)N * 16, BS), BS, 0, stream>>>(
      (const float4*)xw, selfinv, (const float4*)bls, (float4*)out_ls, (long long)N * 16);
  vgae_gcn_scatter4<<<vgae_nblk((long long)E * 16, BS), BS, 0, stream>>>(
      (const float4*)xw, src32, dst32, wnorm, out_ls, E);
}
